// FP8Linear_65154653880606
// MI455X (gfx1250) — 8-bit matrix operand order repaired and checked by reading, compiled, not run on hardware
//
#include <hip/hip_runtime.h>
#include <stdint.h>


typedef int v8i __attribute__((ext_vector_type(8)));

__global__ void fp8lin_absmax(const float4* __restrict__ x, long long n4,
                              unsigned* __restrict__ out_bits) {
    __shared__ float smem[256];
    float m = 0.0f;
    long long i = (long long)blockIdx.x * blockDim.x + threadIdx.x;
    const long long stride = (long long)gridDim.x * blockDim.x;
    for (; i < n4; i += stride) {
        float4 v = x[i];
        m = fmaxf(m, fmaxf(fmaxf(fabsf(v.x), fabsf(v.y)),
                           fmaxf(fabsf(v.z), fabsf(v.w))));
    }
    smem[threadIdx.x] = m;
    __syncthreads();
    #pragma unroll
    for (int s = 128; s > 0; s >>= 1) {
        if ((int)threadIdx.x < s)
            smem[threadIdx.x] = fmaxf(smem[threadIdx.x], smem[threadIdx.x + s]);
        __syncthreads();
    }
    if (threadIdx.x == 0)
        atomicMax(out_bits, __float_as_uint(smem[0]));
}

__device__ __forceinline__ unsigned pack4(float4 v, float s) {
    int c0 = (int)rintf(v.x * s); c0 = min(127, max(-127, c0));
    int c1 = (int)rintf(v.y * s); c1 = min(127, max(-127, c1));
    int c2 = (int)rintf(v.z * s); c2 = min(127, max(-127, c2));
    int c3 = (int)rintf(v.w * s); c3 = min(127, max(-127, c3));
    return (unsigned)(c0 & 0xFF) | ((unsigned)(c1 & 0xFF) << 8) |
           ((unsigned)(c2 & 0xFF) << 16) | ((unsigned)(c3 & 0xFF) << 24);
}

__global__ void fp8lin_quantA(const float* __restrict__ x,
                              unsigned char* __restrict__ aws,
                              const unsigned* __restrict__ amax_bits, int K) {
    const int ktiles = K >> 6;
    const int t    = blockIdx.x * 8 + (threadIdx.x >> 5);
    const int lane = threadIdx.x & 31;
    const int mt = t / ktiles, kt = t - mt * ktiles;
    const int Mr = lane & 15, half = lane >> 4;

    const float amax = __uint_as_float(*amax_bits);
    const float s = amax > 0.0f ? 127.0f / amax : 0.0f;

    const float* row = x + (long long)(mt * 16 + Mr) * K + kt * 64;
    unsigned w[8];
    #pragma unroll
    for (int g = 0; g < 4; ++g) {
        const float4* p = (const float4*)(row + (g >> 1) * 32 + half * 16 + (g & 1) * 8);
        w[2 * g + 0] = pack4(p[0], s);
        w[2 * g + 1] = pack4(p[1], s);
    }
    uint4* dst = (uint4*)(aws + (size_t)t * 1024 + lane * 32);
    dst[0] = make_uint4(w[0], w[1], w[2], w[3]);
    dst[1] = make_uint4(w[4], w[5], w[6], w[7]);
}

__global__ void fp8lin_quantB(const float* __restrict__ wt,
                              unsigned char* __restrict__ bws,
                              const unsigned* __restrict__ amax_bits, int K) {
    const int ktiles = K >> 6;
    const int t    = blockIdx.x * 8 + (threadIdx.x >> 5);
    const int lane = threadIdx.x & 31;
    const int nt = t / ktiles, kt = t - nt * ktiles;
    const int Nc = lane & 15, halfK = lane >> 4;

    const float amax = __uint_as_float(*amax_bits);
    const float s = amax > 0.0f ? 127.0f / amax : 0.0f;

    const float* row = wt + (long long)(nt * 16 + Nc) * K + kt * 64;
    unsigned w[8];
    #pragma unroll
    for (int c = 0; c < 2; ++c) {
        const float4* p = (const float4*)(row + c * 32 + halfK * 16);
        #pragma unroll
        for (int q = 0; q < 4; ++q) w[c * 4 + q] = pack4(p[q], s);
    }
    uint4* dst = (uint4*)(bws + (size_t)t * 1024 + lane * 32);
    dst[0] = make_uint4(w[0], w[1], w[2], w[3]);
    dst[1] = make_uint4(w[4], w[5], w[6], w[7]);
}

__global__ void __launch_bounds__(256)
fp8lin_gemm_iu8(const unsigned char* __restrict__ aws,
                const unsigned char* __restrict__ bws,
                const float* __restrict__ bias,
                const unsigned* __restrict__ amax_bits,
                float* __restrict__ out, int M, int N, int K) {
    const int ktiles = K >> 6;
    const int wave = threadIdx.x >> 5;
    const int lane = threadIdx.x & 31;
    const int mg = wave >> 2;
    const int ng = wave & 3;
    const int mt0 = blockIdx.y * 8 + mg * 4;
    const int nt0 = blockIdx.x * 16 + ng * 4;

    const v8i* Af[4];
    const v8i* Bf[4];
    #pragma unroll
    for (int i = 0; i < 4; ++i)
        Af[i] = (const v8i*)(aws + (size_t)(mt0 + i) * ktiles * 1024 + lane * 32);
    #pragma unroll
    for (int j = 0; j < 4; ++j)
        Bf[j] = (const v8i*)(bws + (size_t)(nt0 + j) * ktiles * 1024 + lane * 32);

    v8i acc[4][4];
    #pragma unroll
    for (int i = 0; i < 4; ++i)
        #pragma unroll
        for (int j = 0; j < 4; ++j)
            acc[i][j] = (v8i){0, 0, 0, 0, 0, 0, 0, 0};

    #pragma unroll 2
    for (int kt = 0; kt < ktiles; ++kt) {
        v8i a[4], b[4];
        #pragma unroll
        for (int i = 0; i < 4; ++i) a[i] = Af[i][kt * 32];
        #pragma unroll
        for (int j = 0; j < 4; ++j) b[j] = Bf[j][kt * 32];

        __builtin_prefetch((const void*)&Af[0][(kt + 1) * 32], 0, 3);
        __builtin_prefetch((const void*)&Bf[0][(kt + 1) * 32], 0, 3);

        #pragma unroll
        for (int i = 0; i < 4; ++i)
            #pragma unroll
            for (int j = 0; j < 4; ++j)
                acc[i][j] = __builtin_amdgcn_wmma_i32_16x16x64_iu8(
                     true, a[i],  true, b[j],
                    acc[i][j],  false,  false);
    }

    const float ax = __uint_as_float(amax_bits[0]);
    const float aw = __uint_as_float(amax_bits[1]);
    const float factor = ax * aw * (1.0f / 16129.0f);

    #pragma unroll
    for (int i = 0; i < 4; ++i) {
        const int m0 = blockIdx.y * 128 + mg * 64 + i * 16 + (lane >> 4) * 8;
        #pragma unroll
        for (int j = 0; j < 4; ++j) {
            const int n = blockIdx.x * 256 + ng * 64 + j * 16 + (lane & 15);
            const float bv = bias[n];
            #pragma unroll
            for (int v = 0; v < 8; ++v)
                out[(size_t)(m0 + v) * N + n] = (float)acc[i][j][v] * factor + bv;
        }
    }
}

extern "C" void kernel_launch(void* const* d_in, const int* in_sizes, int n_in,
                              void* d_out, int out_size, void* d_ws, size_t ws_size,
                              hipStream_t stream) {
    const float* x    = (const float*)d_in[0];
    const float* wt   = (const float*)d_in[1];
    const float* bias = (const float*)d_in[2];
    float* out = (float*)d_out;

    const int N = in_sizes[2];
    const int K = (int)((long long)in_sizes[1] / N);
    const int M = (int)((long long)in_sizes[0] / K);
    const int ktiles = K / 64;

    unsigned char* ws = (unsigned char*)d_ws;
    unsigned* amax_bits  = (unsigned*)ws;
    unsigned char* aws = ws + 256;
    unsigned char* bws = aws + (size_t)M * K;

    hipMemsetAsync(amax_bits, 0, 8, stream);

    fp8lin_absmax<<<2048, 256, 0, stream>>>((const float4*)x,
                                            (long long)in_sizes[0] / 4, amax_bits + 0);
    fp8lin_absmax<<<2048, 256, 0, stream>>>((const float4*)wt,
                                            (long long)in_sizes[1] / 4, amax_bits + 1);

    const int a_tiles = (M / 16) * ktiles;
    const int b_tiles = (N / 16) * ktiles;
    fp8lin_quantA<<<a_tiles / 8, 256, 0, stream>>>(x,  aws, amax_bits + 0, K);
    fp8lin_quantB<<<b_tiles / 8, 256, 0, stream>>>(wt, bws, amax_bits + 1, K);

    dim3 grid(N / 256, M / 128);
    fp8lin_gemm_iu8<<<grid, 256, 0, stream>>>(aws, bws, bias, amax_bits,
                                              out, M, N, K);
}
